// SuperGlueMultiHeadAttention_19215683682996
// MI455X (gfx1250) — matrix operand order repaired and checked by reading, not compiled, not run on hardware
//
#include <hip/hip_runtime.h>
#include <hip/hip_bf16.h>

typedef __attribute__((ext_vector_type(16))) _Float16 v16h;
typedef __attribute__((ext_vector_type(8)))  _Float16 half8;
typedef __attribute__((ext_vector_type(8)))  float    v8f;

#define DEV __device__ __forceinline__

// ---------- WMMA fragment loaders (CDNA5 16-bit layouts, wave32) ----------
// A (16x32, MxK): lane l: m = l&15, g = l>>4; j<8 -> k = 8g+j ; j>=8 -> k = 16+8g+(j-8)
DEV v16h load_frag_a(const _Float16* p, int ld, int row0, int col0) {
    int lane = threadIdx.x & 31;
    int m = lane & 15, g = lane >> 4;
    const _Float16* r = p + (size_t)(row0 + m) * ld + col0 + 8 * g;
    half8 lo = *(const half8*)r;
    half8 hi = *(const half8*)(r + 16);
    v16h out;
#pragma unroll
    for (int j = 0; j < 8; ++j) { out[j] = lo[j]; out[j + 8] = hi[j]; }
    return out;
}

// B (32x16, KxN) loaded from K-contiguous ("transposed") storage S[n][k], row stride ld:
// lane l: n = l&15, g = l>>4; j<8 -> k = 8g+j ; j>=8 -> k = 16+8g+(j-8)
DEV v16h load_frag_bT(const _Float16* S, int ld, int n0, int k0) {
    int lane = threadIdx.x & 31;
    int n = lane & 15, g = lane >> 4;
    const _Float16* r = S + (size_t)(n0 + n) * ld + k0 + 8 * g;
    half8 lo = *(const half8*)r;
    half8 hi = *(const half8*)(r + 16);
    v16h out;
#pragma unroll
    for (int j = 0; j < 8; ++j) { out[j] = lo[j]; out[j + 8] = hi[j]; }
    return out;
}

DEV v8f wmma_f16(v16h a, v16h b, v8f c) {
    return __builtin_amdgcn_wmma_f32_16x16x32_f16(false, a, false, b, (short)0, c, false, false);
}

// ---------- CDNA5 async global->LDS copy (ASYNCcnt-tracked) ----------
DEV void async_ld_b128(void* lds_ptr, const void* gptr) {
    unsigned int lo = (unsigned int)(unsigned long long)lds_ptr;  // LDS byte offset (low 32 bits)
    asm volatile("global_load_async_to_lds_b128 %0, %1, off"
                 :: "v"(lo), "v"(gptr)
                 : "memory");
}
DEV void wait_async_le16() { asm volatile("s_wait_asynccnt 0x10" ::: "memory"); }

// ---------- prep kernels ----------
__global__ void convert_weights_k(const float* wq, const float* wk, const float* wv,
                                  const float* wm, _Float16* out) {
    int i = blockIdx.x * blockDim.x + threadIdx.x;   // 0..262143
    int which = i >> 16, j = i & 65535;
    const float* s = (which == 0) ? wq : (which == 1) ? wk : (which == 2) ? wv : wm;
    out[i] = (_Float16)s[j];
}

// Xt[b][n][c] = (f16) X[b][c][n]   (C=256, N=2048), 32x32 LDS tile transpose
__global__ void transpose_convert_k(const float* X, _Float16* Xt) {
    __shared__ float tile[32][33];
    const int C = 256, N = 2048, cb = C / 32, nb = N / 32;
    int b = blockIdx.x / (cb * nb);
    int rem = blockIdx.x % (cb * nb);
    int ct = rem / nb, nt = rem % nb;
    const float* Xb = X + ((size_t)b * C + ct * 32) * N + nt * 32;
#pragma unroll
    for (int i = 0; i < 32; i += 8)
        tile[threadIdx.y + i][threadIdx.x] = Xb[(size_t)(threadIdx.y + i) * N + threadIdx.x];
    __syncthreads();
    _Float16* Xo = Xt + ((size_t)b * N + nt * 32) * C + ct * 32;
#pragma unroll
    for (int i = 0; i < 32; i += 8)
        Xo[(size_t)(threadIdx.y + i) * C + threadIdx.x] = (_Float16)tile[threadIdx.x][threadIdx.y + i];
}

// ---------- projection GEMM core: acc = W(16x256) * X(256x64) ----------
DEV void proj_tile(const _Float16* __restrict__ W, const _Float16* __restrict__ Xb,
                   int o0, int n0, v8f acc[4]) {
#pragma unroll
    for (int k0 = 0; k0 < 256; k0 += 32) {
        v16h a = load_frag_a(W, 256, o0, k0);
#pragma unroll
        for (int t = 0; t < 4; ++t) {
            v16h bf = load_frag_bT(Xb, 256, n0 + 16 * t, k0);
            acc[t] = wmma_f16(a, bf, acc[t]);
        }
    }
}

// Q/K projections: write per-head layout Hout[b][h][n][64], h = o&3, dd = o>>2
__global__ void proj_qkv_k(const _Float16* __restrict__ W, const _Float16* __restrict__ Xt,
                           const float* __restrict__ bias, _Float16* __restrict__ Hout) {
    const int OT = 16, NT = 32;
    int wave = blockIdx.x * (blockDim.x >> 5) + (threadIdx.x >> 5);
    int b = wave / (OT * NT);
    int rem = wave % (OT * NT);
    int o0 = (rem / NT) * 16, n0 = (rem % NT) * 64;
    v8f acc[4] = {};
    proj_tile(W, Xt + (size_t)b * 2048 * 256, o0, n0, acc);
    int lane = threadIdx.x & 31, col = lane & 15, g = lane >> 4;
#pragma unroll
    for (int t = 0; t < 4; ++t) {
        int n = n0 + 16 * t + col;
#pragma unroll
        for (int r = 0; r < 8; ++r) {
            int o = o0 + r + 8 * g;
            float v = acc[t][r] + bias[o];
            Hout[(((size_t)b * 4 + (o & 3)) * 2048 + n) * 64 + (o >> 2)] = (_Float16)v;
        }
    }
}

// V projection: write TRANSPOSED per-head layout VT[b][h][64][2048]
__global__ void proj_v_k(const _Float16* __restrict__ W, const _Float16* __restrict__ Xt,
                         const float* __restrict__ bias, _Float16* __restrict__ VT) {
    const int OT = 16, NT = 32;
    int wave = blockIdx.x * (blockDim.x >> 5) + (threadIdx.x >> 5);
    int b = wave / (OT * NT);
    int rem = wave % (OT * NT);
    int o0 = (rem / NT) * 16, n0 = (rem % NT) * 64;
    v8f acc[4] = {};
    proj_tile(W, Xt + (size_t)b * 2048 * 256, o0, n0, acc);
    int lane = threadIdx.x & 31, col = lane & 15, g = lane >> 4;
#pragma unroll
    for (int t = 0; t < 4; ++t) {
        int n = n0 + 16 * t + col;
#pragma unroll
        for (int r = 0; r < 8; ++r) {
            int o = o0 + r + 8 * g;
            float v = acc[t][r] + bias[o];
            VT[(((size_t)b * 4 + (o & 3)) * 64 + (o >> 2)) * 2048 + n] = (_Float16)v;
        }
    }
}

// final projection: f32 output [B][256][2048]
__global__ void proj_out_k(const _Float16* __restrict__ W, const _Float16* __restrict__ Xt,
                           const float* __restrict__ bias, float* __restrict__ Out) {
    const int OT = 16, NT = 32;
    int wave = blockIdx.x * (blockDim.x >> 5) + (threadIdx.x >> 5);
    int b = wave / (OT * NT);
    int rem = wave % (OT * NT);
    int o0 = (rem / NT) * 16, n0 = (rem % NT) * 64;
    v8f acc[4] = {};
    proj_tile(W, Xt + (size_t)b * 2048 * 256, o0, n0, acc);
    int lane = threadIdx.x & 31, col = lane & 15, g = lane >> 4;
#pragma unroll
    for (int t = 0; t < 4; ++t) {
        int n = n0 + 16 * t + col;
#pragma unroll
        for (int r = 0; r < 8; ++r) {
            int o = o0 + r + 8 * g;
            Out[((size_t)b * 256 + o) * 2048 + n] = acc[t][r] + bias[o];
        }
    }
}

// ---------- flash attention: block = (b, h, 64 query rows), 4 waves ----------
// Qh/Kh: [B][4][2048][64] f16.  VT: [B][4][64][2048] f16 (pre-transposed).
// Xout: [B][2048][256] f16 with channel c = dd*4 + h.
// Double-buffered async global->LDS staging pipelined against WMMA compute.
__global__ void attn_k(const _Float16* __restrict__ Qh, const _Float16* __restrict__ Kh,
                       const _Float16* __restrict__ VT, _Float16* __restrict__ Xout) {
    __shared__ __align__(16) _Float16 Kt[2][64 * 64];   // [m][dd], double-buffered
    __shared__ __align__(16) _Float16 Vt[2][64 * 64];   // [dd][m], double-buffered
    __shared__ __align__(16) _Float16 Ps[4][16 * 64];   // per-wave P scratch
    int wid = threadIdx.x >> 5, lane = threadIdx.x & 31;
    int col = lane & 15, g = lane >> 4;

    const int nb = 32;  // 2048/64
    int b = blockIdx.x / (4 * nb);
    int rem = blockIdx.x % (4 * nb);
    int h = rem / nb, qt = rem % nb;
    int q0 = qt * 64 + wid * 16;

    const _Float16* Qp = Qh + (((size_t)b * 4 + h) * 2048) * 64;
    const _Float16* Kp = Kh + (((size_t)b * 4 + h) * 2048) * 64;
    const _Float16* Vp = VT + (((size_t)b * 4 + h) * 64) * 2048;

    // per-thread staging coordinates (16B chunks, 8 per tile per tensor)
    int strow[8], stcx[8];
#pragma unroll
    for (int i = 0; i < 8; ++i) {
        int lin = threadIdx.x + 128 * i;   // 0..1023
        strow[i] = lin >> 3;               // 128 rows, 128B each
        stcx[i] = (lin & 7) * 8;           // halves offset
    }

    v16h qa0 = load_frag_a(Qp, 64, q0, 0);
    v16h qa1 = load_frag_a(Qp, 64, q0, 32);
    v16h ones;
#pragma unroll
    for (int j = 0; j < 16; ++j) ones[j] = (_Float16)1.0f;

    float M[8];
    v8f acco[4] = {};
    v8f accl = {};
#pragma unroll
    for (int r = 0; r < 8; ++r) M[r] = -__builtin_inff();
    const float sc2 = 0.125f * 1.44269504088896f;  // (1/sqrt(64)) * log2(e)

    // prologue: stage tile 0 (16 async instructions per wave)
#pragma unroll
    for (int i = 0; i < 8; ++i) {
        async_ld_b128(Kt[0] + strow[i] * 64 + stcx[i], Kp + (size_t)strow[i] * 64 + stcx[i]);
        async_ld_b128(Vt[0] + strow[i] * 64 + stcx[i], Vp + (size_t)strow[i] * 2048 + stcx[i]);
    }

    for (int it = 0; it < 32; ++it) {
        int kb = it * 64;
        // issue next tile into the other buffer (wraps to tile 0 on last iter: harmless)
        {
            int nkb = ((it + 1) & 31) * 64;
            _Float16* Kb = Kt[(it + 1) & 1];
            _Float16* Vb = Vt[(it + 1) & 1];
#pragma unroll
            for (int i = 0; i < 8; ++i) {
                async_ld_b128(Kb + strow[i] * 64 + stcx[i],
                              Kp + (size_t)(nkb + strow[i]) * 64 + stcx[i]);
                async_ld_b128(Vb + strow[i] * 64 + stcx[i],
                              Vp + (size_t)strow[i] * 2048 + nkb + stcx[i]);
            }
        }
        wait_async_le16();   // current tile's 16 copies complete (in-order)
        __syncthreads();     // visible to all waves
        const _Float16* Kc = Kt[it & 1];
        const _Float16* Vc = Vt[it & 1];

        // scores: 4 key sub-tiles x 2 k-steps (d=64)
        v8f s0 = {}, s1 = {}, s2 = {}, s3 = {};
        s0 = wmma_f16(qa0, load_frag_bT(Kc, 64, 0, 0), s0);
        s0 = wmma_f16(qa1, load_frag_bT(Kc, 64, 0, 32), s0);
        s1 = wmma_f16(qa0, load_frag_bT(Kc, 64, 16, 0), s1);
        s1 = wmma_f16(qa1, load_frag_bT(Kc, 64, 16, 32), s1);
        s2 = wmma_f16(qa0, load_frag_bT(Kc, 64, 32, 0), s2);
        s2 = wmma_f16(qa1, load_frag_bT(Kc, 64, 32, 32), s2);
        s3 = wmma_f16(qa0, load_frag_bT(Kc, 64, 48, 0), s3);
        s3 = wmma_f16(qa1, load_frag_bT(Kc, 64, 48, 32), s3);

        // online softmax in log2 domain; row = r + 8g
        float p0[8], p1[8], p2[8], p3[8];
#pragma unroll
        for (int r = 0; r < 8; ++r) {
            float v0 = s0[r] * sc2, v1 = s1[r] * sc2, v2 = s2[r] * sc2, v3 = s3[r] * sc2;
            float mx = fmaxf(fmaxf(v0, v1), fmaxf(v2, v3));
#pragma unroll
            for (int d = 1; d < 16; d <<= 1) mx = fmaxf(mx, __shfl_xor(mx, d, 32));
            float mt = fmaxf(M[r], mx);
            float alpha = exp2f(M[r] - mt);
            M[r] = mt;
            p0[r] = exp2f(v0 - mt); p1[r] = exp2f(v1 - mt);
            p2[r] = exp2f(v2 - mt); p3[r] = exp2f(v3 - mt);
#pragma unroll
            for (int c = 0; c < 4; ++c) acco[c][r] *= alpha;
            accl[r] *= alpha;
        }

        // P: C-layout -> A-layout via wave-private LDS (16x64)
        _Float16* Pw = Ps[wid];
#pragma unroll
        for (int r = 0; r < 8; ++r) {
            int ro = (r + 8 * g) * 64;
            Pw[ro + col]      = (_Float16)p0[r];
            Pw[ro + 16 + col] = (_Float16)p1[r];
            Pw[ro + 32 + col] = (_Float16)p2[r];
            Pw[ro + 48 + col] = (_Float16)p3[r];
        }
        v16h pa0 = load_frag_a(Pw, 64, 0, 0);
        v16h pa1 = load_frag_a(Pw, 64, 0, 32);
#pragma unroll
        for (int c = 0; c < 4; ++c) {
            acco[c] = wmma_f16(pa0, load_frag_bT(Vc, 64, c * 16, 0), acco[c]);
            acco[c] = wmma_f16(pa1, load_frag_bT(Vc, 64, c * 16, 32), acco[c]);
        }
        // row-sum of P via WMMA with ones (denominator accumulator)
        accl = wmma_f16(pa0, ones, accl);
        accl = wmma_f16(pa1, ones, accl);

        __syncthreads();   // all waves done reading buf[it&1] before it's refilled
    }

    // normalize + store, interleaved-head channel c = dd*4 + h
#pragma unroll
    for (int c = 0; c < 4; ++c) {
        int dd = c * 16 + col;
        int ch = dd * 4 + h;
#pragma unroll
        for (int r = 0; r < 8; ++r) {
            int n = q0 + r + 8 * g;
            float v = acco[c][r] / accl[r];
            Xout[((size_t)b * 2048 + n) * 256 + ch] = (_Float16)v;
        }
    }
}

extern "C" void kernel_launch(void* const* d_in, const int* in_sizes, int n_in,
                              void* d_out, int out_size, void* d_ws, size_t ws_size,
                              hipStream_t stream) {
    const float* query = (const float*)d_in[0];
    const float* key   = (const float*)d_in[1];
    const float* value = (const float*)d_in[2];
    const float* w_q = (const float*)d_in[3];
    const float* b_q = (const float*)d_in[4];
    const float* w_k = (const float*)d_in[5];
    const float* b_k = (const float*)d_in[6];
    const float* w_v = (const float*)d_in[7];
    const float* b_v = (const float*)d_in[8];
    const float* w_m = (const float*)d_in[9];
    const float* b_m = (const float*)d_in[10];
    float* out = (float*)d_out;

    char* ws = (char*)d_ws;
    const size_t W16_B = 4 * 65536 * sizeof(_Float16);               // 512 KB
    const size_t TEN_B = (size_t)8 * 2048 * 256 * sizeof(_Float16);  // 8 MB
    _Float16* W16 = (_Float16*)ws;
    _Float16* XTq = (_Float16*)(ws + W16_B);
    _Float16* XTk = (_Float16*)(ws + W16_B + TEN_B);
    _Float16* XTv = (_Float16*)(ws + W16_B + 2 * TEN_B);
    _Float16* QH  = (_Float16*)(ws + W16_B + 3 * TEN_B);
    _Float16* KH  = (_Float16*)(ws + W16_B + 4 * TEN_B);
    _Float16* VT  = (_Float16*)(ws + W16_B + 5 * TEN_B);
    _Float16* XA  = (_Float16*)(ws + W16_B + 6 * TEN_B);

    convert_weights_k<<<1024, 256, 0, stream>>>(w_q, w_k, w_v, w_m, W16);
    transpose_convert_k<<<4096, dim3(32, 8), 0, stream>>>(query, XTq);
    transpose_convert_k<<<4096, dim3(32, 8), 0, stream>>>(key, XTk);
    transpose_convert_k<<<4096, dim3(32, 8), 0, stream>>>(value, XTv);

    proj_qkv_k<<<1024, 128, 0, stream>>>(W16 + 0 * 65536, XTq, b_q, QH);
    proj_qkv_k<<<1024, 128, 0, stream>>>(W16 + 1 * 65536, XTk, b_k, KH);
    proj_v_k  <<<1024, 128, 0, stream>>>(W16 + 2 * 65536, XTv, b_v, VT);

    attn_k<<<1024, 128, 0, stream>>>(QH, KH, VT, XA);

    proj_out_k<<<1024, 128, 0, stream>>>(W16 + 3 * 65536, XA, b_m, out);
}
